// MultiHeadAttention_22900765622419
// MI455X (gfx1250) — matrix operand order repaired and checked by reading, not compiled, not run on hardware
//
#include <hip/hip_runtime.h>
#include <hip/hip_bf16.h>

// ---------------------------------------------------------------------------
// MultiHeadAttention for MI455X (gfx1250): bf16 WMMA pipeline with
// async-to-LDS double-buffered staging of shared tiles.
//   1) convert x, Wq/Wk/Wv/Wo  f32 -> bf16
//   2) Q = x@Wq^T+bq -> [B,H,N,DK];  K likewise;  V -> [B,H,DK,N] (transposed)
//   3) flash attention per (b,h): S = (Q K^T)/8 * mask, online softmax,
//      O += P@V; K/V tiles async-staged in LDS; P transposed via LDS
//   4) out = concat@Wo^T + bo -> f32 d_out
// ---------------------------------------------------------------------------

#define BB  2
#define NN  2048
#define MM  1024
#define HH  16
#define DKK 64
#define ROWS (BB * NN)   // 4096 GEMM rows

typedef __bf16 bf16;
typedef __attribute__((ext_vector_type(16))) __bf16 v16bf;
typedef __attribute__((ext_vector_type(8)))  __bf16 v8bf;
typedef __attribute__((ext_vector_type(8)))  float  v8f;

static __device__ inline v8f wmma_bf16(v16bf a, v16bf b, v8f c) {
  return __builtin_amdgcn_wmma_f32_16x16x32_bf16(
      false, a, false, b, (short)0, c, false, false);
}

// A-matrix 16x32 bf16 fragment: lane holds row (lane&15);
// VGPR0..3 -> K = 8*half + 0..7, VGPR4..7 -> K = 16 + 8*half + 0..7.
static __device__ inline v16bf load_frag_a(const bf16* p, int ld) {
  int lane = threadIdx.x & 31;
  int row = lane & 15, half = lane >> 4;
  const bf16* b = p + row * ld + half * 8;
  union { v16bf v; v8bf h[2]; } u;
  u.h[0] = *(const v8bf*)(b);
  u.h[1] = *(const v8bf*)(b + 16);
  return u.v;
}

// B-matrix 32x16 bf16 fragment: lane holds column (lane&15);
// VGPR0..3 -> K = 8*half + 0..7, VGPR4..7 -> K = 16 + 8*half + 0..7.  Source row (lane&15), stride ld.
static __device__ inline v16bf load_frag_b(const bf16* p, int ld) {
  int lane = threadIdx.x & 31;
  int col = lane & 15, half = lane >> 4;
  const bf16* b = p + col * ld + half * 8;
  union { v16bf v; v8bf h[2]; } u;
  u.h[0] = *(const v8bf*)(b);
  u.h[1] = *(const v8bf*)(b + 16);
  return u.v;
}

// ---- CDNA5 async global->LDS copy (16B per lane), tracked by ASYNCcnt ------
static __device__ inline uint32_t lds_addr_of(const void* p) {
  // LDS flat aperture keeps the byte offset in addr[31:0] (ISA 10.2)
  return (uint32_t)(size_t)p;
}
static __device__ inline void async_b128(uint32_t lds_dst, const bf16* g) {
  asm volatile("global_load_async_to_lds_b128 %0, %1, off"
               :: "v"(lds_dst), "v"((unsigned long long)(size_t)g)
               : "memory");
}
#define WAIT_ASYNC(n) asm volatile("s_wait_asynccnt " #n ::: "memory")
#define WAIT_DS0()    asm volatile("s_wait_dscnt 0x0" ::: "memory")

// ---------------------------------------------------------------------------
__global__ __launch_bounds__(256) void cvt_f32_bf16(const float* __restrict__ src,
                                                    bf16* __restrict__ dst, int n) {
  int i = (blockIdx.x * blockDim.x + threadIdx.x) * 4;
  if (i + 3 < n) {
    float4 f = *(const float4*)(src + i);
    dst[i + 0] = (bf16)f.x; dst[i + 1] = (bf16)f.y;
    dst[i + 2] = (bf16)f.z; dst[i + 3] = (bf16)f.w;
  }
}

// ---------------------------------------------------------------------------
// GEMM: Y[row,col] = sum_k A[row,k]*W[col,k] + bias[col]
// W tile (64 cols x 32 k, 4KB) async-staged to LDS, double buffered,
// shared by all 8 waves.  Wave computes 32 rows x 64 cols (8 WMMA / k-step).
// grid: (MM/64, ROWS/256), block 256.
// mode 0: bf16 out [B,H,N,DK];  mode 1: bf16 out [B,H,DK,N];  mode 2: f32 row-major
// ---------------------------------------------------------------------------
__global__ __launch_bounds__(256) void gemm_wmma(const bf16* __restrict__ A,
                                                 const bf16* __restrict__ W,
                                                 const float* __restrict__ bias,
                                                 bf16* __restrict__ outb,
                                                 float* __restrict__ outf,
                                                 int mode) {
  __shared__ bf16 wbuf[2][64 * 32];   // [col][k], 4KB x 2

  const int tid  = threadIdx.x;
  const int wave = tid >> 5, lane = tid & 31;
  const int row0 = blockIdx.y * 256 + wave * 32;
  const int col0 = blockIdx.x * 64;

  // cooperative W staging: thread -> (col, 8-elem segment)
  const int wcol = tid >> 2, wseg = tid & 3;
  const bf16* wsrc = W + (size_t)(col0 + wcol) * MM + wseg * 8;
  const uint32_t wdst[2] = { lds_addr_of(&wbuf[0][wcol * 32 + wseg * 8]),
                             lds_addr_of(&wbuf[1][wcol * 32 + wseg * 8]) };

  async_b128(wdst[0], wsrc);                       // k = 0 tile

  v8f acc[2][4] = {};
  v16bf a0 = load_frag_a(A + (size_t)row0 * MM, MM);
  v16bf a1 = load_frag_a(A + (size_t)(row0 + 16) * MM, MM);

  for (int ki = 0; ki < MM / 32; ++ki) {
    const int k = ki * 32;
    v16bf na0 = a0, na1 = a1;
    if (ki + 1 < MM / 32) {
      async_b128(wdst[(ki + 1) & 1], wsrc + k + 32);  // next W tile
      na0 = load_frag_a(A + (size_t)row0 * MM + k + 32, MM);
      na1 = load_frag_a(A + (size_t)(row0 + 16) * MM + k + 32, MM);
      WAIT_ASYNC(0x1);                               // current tile landed
    } else {
      WAIT_ASYNC(0x0);
    }
    __syncthreads();

    const bf16* wb = wbuf[ki & 1];
#pragma unroll
    for (int t = 0; t < 4; ++t) {
      v16bf b = load_frag_b(wb + 16 * t * 32, 32);
      acc[0][t] = wmma_bf16(a0, b, acc[0][t]);
      acc[1][t] = wmma_bf16(a1, b, acc[1][t]);
    }
    __syncthreads();
    a0 = na0; a1 = na1;
  }

  const int half = lane >> 4, nIdx = lane & 15;
#pragma unroll
  for (int i = 0; i < 2; ++i) {
#pragma unroll
    for (int t = 0; t < 4; ++t) {
      int col = col0 + 16 * t + nIdx;
      float bv = bias[col];
#pragma unroll
      for (int r = 0; r < 8; ++r) {
        int row = row0 + 16 * i + r + 8 * half;
        float v = acc[i][t][r] + bv;
        if (mode == 2) {
          outf[(size_t)row * MM + col] = v;
        } else {
          int b = row >> 11, n = row & (NN - 1);
          int h = col >> 6,  d = col & (DKK - 1);
          if (mode == 0)
            outb[(((size_t)(b * HH + h) * NN) + n) * DKK + d] = (bf16)v;
          else
            outb[(((size_t)(b * HH + h) * DKK) + d) * NN + n] = (bf16)v;
        }
      }
    }
  }
}

// ---------------------------------------------------------------------------
// Flash attention per (b,h). grid: (N/128, B*H), block 256 (8 waves).
// K tile (32 keys x 64 d) and V tile (64 d x 32 keys) async-staged to LDS,
// double buffered, shared by all 8 waves.  Wave owns 16 query rows.
// ---------------------------------------------------------------------------
__global__ __launch_bounds__(256) void attn_wmma(const bf16* __restrict__ Q,
                                                 const bf16* __restrict__ Km,
                                                 const bf16* __restrict__ Vt,
                                                 const float* __restrict__ mask,
                                                 bf16* __restrict__ concat) {
  __shared__ bf16 kbuf[2][32 * 64];   // [key][d]
  __shared__ bf16 vbuf[2][64 * 32];   // [d][key]
  __shared__ bf16 pbuf[8][16 * 32];   // per-wave P tile

  const int tid  = threadIdx.x;
  const int wave = tid >> 5, lane = tid & 31;
  const int half = lane >> 4, nIdx = lane & 15;

  const int bh = blockIdx.y;
  const int b  = bh >> 4, h = bh & (HH - 1);
  const int q0 = blockIdx.x * 128 + wave * 16;

  const bf16* Qp = Q  + ((size_t)bh * NN + q0) * DKK;
  const bf16* Kp = Km + (size_t)bh * NN * DKK;
  const bf16* Vp = Vt + (size_t)bh * DKK * NN;
  const float* Mp = mask + (size_t)b * NN * NN;

  // cooperative staging maps (one 16B async chunk per thread per tile)
  const int kkey = tid >> 3, kseg = tid & 7;            // K: 32x64
  const bf16* ksrc = Kp + (size_t)kkey * DKK + kseg * 8;
  const uint32_t kdst[2] = { lds_addr_of(&kbuf[0][kkey * 64 + kseg * 8]),
                             lds_addr_of(&kbuf[1][kkey * 64 + kseg * 8]) };
  const int vd = tid >> 2, vseg = tid & 3;              // V: 64x32
  const bf16* vsrc = Vp + (size_t)vd * NN + vseg * 8;
  const uint32_t vdst[2] = { lds_addr_of(&vbuf[0][vd * 32 + vseg * 8]),
                             lds_addr_of(&vbuf[1][vd * 32 + vseg * 8]) };

  async_b128(kdst[0], ksrc);                            // kb = 0
  async_b128(vdst[0], vsrc);

  v16bf qa0 = load_frag_a(Qp, DKK);
  v16bf qa1 = load_frag_a(Qp + 32, DKK);

  v8f o[4] = {};
  float mrow[8], lrow[8];
#pragma unroll
  for (int r = 0; r < 8; ++r) { mrow[r] = -1e30f; lrow[r] = 0.f; }

  for (int it = 0; it < NN / 32; ++it) {
    const int kb = it * 32;
    if (it + 1 < NN / 32) {
      async_b128(kdst[(it + 1) & 1], ksrc + (size_t)(kb + 32) * DKK);
      async_b128(vdst[(it + 1) & 1], vsrc + kb + 32);
      WAIT_ASYNC(0x2);                                  // current K+V landed
    } else {
      WAIT_ASYNC(0x0);
    }
    __syncthreads();

    const bf16* kt = kbuf[it & 1];
    const bf16* vt = vbuf[it & 1];

    // ---- scores: two 16x16 tiles, contraction over DK=64
    v8f s[2];
#pragma unroll
    for (int j = 0; j < 2; ++j) {
      v16bf b0 = load_frag_b(kt + 16 * j * 64, 64);
      v16bf b1 = load_frag_b(kt + 16 * j * 64 + 32, 64);
      v8f t = v8f{};
      t = wmma_bf16(qa0, b0, t);
      t = wmma_bf16(qa1, b1, t);
      s[j] = t;
    }

    // ---- scale + multiplicative mask (pre-softmax)
#pragma unroll
    for (int j = 0; j < 2; ++j) {
      int kcol = kb + 16 * j + nIdx;
#pragma unroll
      for (int r = 0; r < 8; ++r) {
        int qrow = q0 + r + 8 * half;
        s[j][r] = s[j][r] * 0.125f * Mp[(size_t)qrow * NN + kcol];
      }
    }

    // ---- online softmax
    float rmax[8];
#pragma unroll
    for (int r = 0; r < 8; ++r) rmax[r] = fmaxf(s[0][r], s[1][r]);
#pragma unroll
    for (int msk = 1; msk <= 8; msk <<= 1)
#pragma unroll
      for (int r = 0; r < 8; ++r)
        rmax[r] = fmaxf(rmax[r], __shfl_xor(rmax[r], msk, 32));

    float alpha[8];
#pragma unroll
    for (int r = 0; r < 8; ++r) {
      float mn = fmaxf(mrow[r], rmax[r]);
      alpha[r] = __expf(mrow[r] - mn);
      mrow[r] = mn;
    }

    float rsum[8];
#pragma unroll
    for (int r = 0; r < 8; ++r) rsum[r] = 0.f;
#pragma unroll
    for (int j = 0; j < 2; ++j)
#pragma unroll
      for (int r = 0; r < 8; ++r) {
        float p = __expf(s[j][r] - mrow[r]);
        s[j][r] = p;
        rsum[r] += p;
      }
#pragma unroll
    for (int msk = 1; msk <= 8; msk <<= 1)
#pragma unroll
      for (int r = 0; r < 8; ++r)
        rsum[r] += __shfl_xor(rsum[r], msk, 32);
#pragma unroll
    for (int r = 0; r < 8; ++r) lrow[r] = lrow[r] * alpha[r] + rsum[r];

#pragma unroll
    for (int t = 0; t < 4; ++t)
#pragma unroll
      for (int r = 0; r < 8; ++r) o[t][r] *= alpha[r];

    // ---- P (C-layout) -> LDS -> A-fragment (wave-private tile)
    bf16* pl = pbuf[wave];
#pragma unroll
    for (int j = 0; j < 2; ++j)
#pragma unroll
      for (int r = 0; r < 8; ++r)
        pl[(r + 8 * half) * 32 + 16 * j + nIdx] = (bf16)s[j][r];
    WAIT_DS0();                                        // wave-local ordering

    v16bf pa = load_frag_a(pl, 32);
#pragma unroll
    for (int t = 0; t < 4; ++t) {
      v16bf vb = load_frag_b(vt + 16 * t * 32, 32);
      o[t] = wmma_bf16(pa, vb, o[t]);
    }
    __syncthreads();                                   // before buffer reuse
  }

  // ---- normalize, write concat[b][n][h*DK + d]
#pragma unroll
  for (int t = 0; t < 4; ++t)
#pragma unroll
    for (int r = 0; r < 8; ++r) {
      int qrow = q0 + r + 8 * half;
      float inv = (lrow[r] != 0.f) ? 1.f / lrow[r] : 0.f;
      concat[((size_t)b * NN + qrow) * MM + h * DKK + 16 * t + nIdx] =
          (bf16)(o[t][r] * inv);
    }
}

// ---------------------------------------------------------------------------
extern "C" void kernel_launch(void* const* d_in, const int* in_sizes, int n_in,
                              void* d_out, int out_size, void* d_ws, size_t ws_size,
                              hipStream_t stream) {
  const float* x    = (const float*)d_in[0];
  const float* mask = (const float*)d_in[1];
  const float* Wq   = (const float*)d_in[2];
  const float* bq   = (const float*)d_in[3];
  const float* Wk   = (const float*)d_in[4];
  const float* bk   = (const float*)d_in[5];
  const float* Wv   = (const float*)d_in[6];
  const float* bv   = (const float*)d_in[7];
  const float* Wo   = (const float*)d_in[8];
  const float* bo   = (const float*)d_in[9];
  float* out = (float*)d_out;

  bf16* ws = (bf16*)d_ws;
  size_t off = 0;
  bf16* xb   = ws + off; off += (size_t)ROWS * MM;
  bf16* Wqb  = ws + off; off += (size_t)MM * MM;
  bf16* Wkb  = ws + off; off += (size_t)MM * MM;
  bf16* Wvb  = ws + off; off += (size_t)MM * MM;
  bf16* Wob  = ws + off; off += (size_t)MM * MM;
  bf16* Qm   = ws + off; off += (size_t)ROWS * MM;   // [B,H,N,DK]
  bf16* Km   = ws + off; off += (size_t)ROWS * MM;   // [B,H,N,DK]
  bf16* Vtm  = ws + off; off += (size_t)ROWS * MM;   // [B,H,DK,N]
  bf16* conc = ws + off; off += (size_t)ROWS * MM;   // [B,N,M]

  {
    int n = ROWS * MM;
    cvt_f32_bf16<<<(n / 4 + 255) / 256, 256, 0, stream>>>(x, xb, n);
    n = MM * MM;
    cvt_f32_bf16<<<(n / 4 + 255) / 256, 256, 0, stream>>>(Wq, Wqb, n);
    cvt_f32_bf16<<<(n / 4 + 255) / 256, 256, 0, stream>>>(Wk, Wkb, n);
    cvt_f32_bf16<<<(n / 4 + 255) / 256, 256, 0, stream>>>(Wv, Wvb, n);
    cvt_f32_bf16<<<(n / 4 + 255) / 256, 256, 0, stream>>>(Wo, Wob, n);
  }

  dim3 gg(MM / 64, ROWS / 256);  // 16 x 16
  gemm_wmma<<<gg, 256, 0, stream>>>(xb, Wqb, bq, Qm,  nullptr, 0);
  gemm_wmma<<<gg, 256, 0, stream>>>(xb, Wkb, bk, Km,  nullptr, 0);
  gemm_wmma<<<gg, 256, 0, stream>>>(xb, Wvb, bv, Vtm, nullptr, 1);

  dim3 ga(NN / 128, BB * HH);    // 16 x 32
  attn_wmma<<<ga, 256, 0, stream>>>(Qm, Km, Vtm, mask, conc);

  gemm_wmma<<<gg, 256, 0, stream>>>(conc, Wob, bo, nullptr, out, 2);
}
